// LSTM_Turnoverv0_53850299957978
// MI455X (gfx1250) — matrix operand order repaired and checked by reading, not compiled, not run on hardware
//
#include <hip/hip_runtime.h>
#include <hip/hip_bf16.h>
#include <math.h>

// ---------------------------------------------------------------------------
// 2-layer LSTM, B=512, T=256, H=1024, fused persistent kernel for gfx1250.
// fp16 WMMA (v_wmma_f32_16x16x32_f16); weights converted to fp16 once per
// call (24MB fp16 -> resident in the 192MB L2). 128 blocks x 256 threads
// (8 wave32) = 1024 waves; each wave owns TWO m-tiles x {i,f,g,o} gate tiles
// (8 accumulators) so every B fragment feeds 2 WMMAs (halves L2 traffic) and
// the k-loop is software-pipelined (double-buffered A/B fragments).
// A (activations) staged to LDS via GLOBAL_LOAD_ASYNC_TO_LDS_B128 (ASYNCcnt).
// Device-wide generation barrier between phases; h-state double buffered.
// ---------------------------------------------------------------------------

typedef _Float16 half_t;
typedef __attribute__((ext_vector_type(16))) _Float16 v16h;
typedef __attribute__((ext_vector_type(8)))  float    v8f;
typedef __attribute__((ext_vector_type(4)))  int      v4i;

#define Hdim 1024
#define Tlen 256
#define Bsz  512
#define NBLK 128
#define LDS_PAD 8
#define LDS_ROW (Hdim + LDS_PAD)        // halves; 2064B stride spreads banks
#define LDS_BYTES (32 * LDS_ROW * 2)    // 32 activation rows, 66048 B

#if __has_builtin(__builtin_amdgcn_global_load_async_to_lds_b128)
#define HAVE_ASYNC_LDS 1
#endif

union AF { v16h v; uint4 q[2]; };
union F8 { v8f  v; float f[8]; };

__device__ __forceinline__ float sigmoidf_(float v) {
    return 1.0f / (1.0f + __expf(-v));
}

// Stage 32 rows x 1024 halves into LDS. 256 threads: thread = row*8 + seg,
// each moves 256 contiguous bytes (16 x b128).
__device__ __forceinline__ void stage_rows32(half_t* lds,
                                             const half_t* __restrict__ src,
                                             int mbase) {
    const int tid = threadIdx.x;
    const int row = tid >> 3;       // 0..31
    const int seg = tid & 7;        // 0..7
    const half_t* s = src + (size_t)(mbase + row) * Hdim + seg * 128;
    half_t*       d = lds + row * LDS_ROW + seg * 128;
#if defined(HAVE_ASYNC_LDS)
#pragma unroll
    for (int c = 0; c < 16; ++c)
        __builtin_amdgcn_global_load_async_to_lds_b128(
            (v4i*)(void*)(s + c * 8), (v4i*)(void*)(d + c * 8), 0, 0);
#else
#pragma unroll
    for (int c = 0; c < 16; ++c)
        ((uint4*)d)[c] = ((const uint4*)s)[c];
#endif
}

__device__ __forceinline__ void stage_wait() {
#if defined(HAVE_ASYNC_LDS)
#if __has_builtin(__builtin_amdgcn_s_wait_asynccnt)
    __builtin_amdgcn_s_wait_asynccnt(0);
#else
    asm volatile("s_wait_asynccnt 0x0" ::: "memory");
#endif
#endif
    __syncthreads();
}

// A fragment (16-bit A 16x32): lane holds row m, 16 consecutive K at
// kb*32+hi*16..   (base already lane-adjusted)
__device__ __forceinline__ void loadA(AF& a, const half_t* base, int kb) {
    const half_t* p = base + kb * 32;
    a.q[0] = *(const uint4*)(p);
    a.q[1] = *(const uint4*)(p + 8);
}
// B fragment (16-bit B 32x16): lane holds col n, 16 consecutive K at hi*16.
__device__ __forceinline__ void loadB(AF& b, const half_t* base, int kb) {
    const half_t* p = base + kb * 32;
    b.q[0] = *(const uint4*)(p);
    b.q[1] = *(const uint4*)(p + 8);
}

#define WMMA(A, B, C) __builtin_amdgcn_wmma_f32_16x16x32_f16( \
    false, (A).v, false, (B).v, (short)0, (C), false, false)

// K=1024 GEMM accumulating 2 m-tiles x 4 gate tiles, software-pipelined.
__device__ __forceinline__ void gemm8(v8f acc[8], const half_t* lds,
                                      const half_t* __restrict__ W, int htile) {
    const int lane = threadIdx.x & 31;
    const int hi   = lane >> 4;
    const int col  = lane & 15;

    const half_t* b0r = W + (size_t)((0 << 10) + (htile << 4) + col) * Hdim + (hi << 4);
    const half_t* b1r = W + (size_t)((1 << 10) + (htile << 4) + col) * Hdim + (hi << 4);
    const half_t* b2r = W + (size_t)((2 << 10) + (htile << 4) + col) * Hdim + (hi << 4);
    const half_t* b3r = W + (size_t)((3 << 10) + (htile << 4) + col) * Hdim + (hi << 4);
    const half_t* a0r = lds + col * LDS_ROW + (hi << 4);
    const half_t* a1r = a0r + 16 * LDS_ROW;

    AF a0c, a1c, b0c, b1c, b2c, b3c;
    loadA(a0c, a0r, 0); loadA(a1c, a1r, 0);
    loadB(b0c, b0r, 0); loadB(b1c, b1r, 0);
    loadB(b2c, b2r, 0); loadB(b3c, b3r, 0);

    for (int kb = 0; kb < 32; ++kb) {
        const int kn = (kb + 1) & 31;          // branchless wrap (extra
        AF a0n, a1n, b0n, b1n, b2n, b3n;       //  redundant load on last iter)
        loadA(a0n, a0r, kn); loadA(a1n, a1r, kn);
        loadB(b0n, b0r, kn); loadB(b1n, b1r, kn);
        loadB(b2n, b2r, kn); loadB(b3n, b3r, kn);

        acc[0] = WMMA(a0c, b0c, acc[0]);
        acc[4] = WMMA(a1c, b0c, acc[4]);
        acc[1] = WMMA(a0c, b1c, acc[1]);
        acc[5] = WMMA(a1c, b1c, acc[5]);
        acc[2] = WMMA(a0c, b2c, acc[2]);
        acc[6] = WMMA(a1c, b2c, acc[6]);
        acc[3] = WMMA(a0c, b3c, acc[3]);
        acc[7] = WMMA(a1c, b3c, acc[7]);

        a0c = a0n; a1c = a1n;
        b0c = b0n; b1c = b1n; b2c = b2n; b3c = b3n;
    }
}

// Device-wide barrier: generation-counted arrive/spin at agent scope.
__device__ __forceinline__ void grid_sync(unsigned* cnt, unsigned* gen) {
    __threadfence();
    __syncthreads();
    if (threadIdx.x == 0) {
        unsigned g = __hip_atomic_load(gen, __ATOMIC_ACQUIRE, __HIP_MEMORY_SCOPE_AGENT);
        unsigned a = __hip_atomic_fetch_add(cnt, 1u, __ATOMIC_ACQ_REL, __HIP_MEMORY_SCOPE_AGENT);
        if (a == NBLK - 1) {
            __hip_atomic_store(cnt, 0u, __ATOMIC_RELAXED, __HIP_MEMORY_SCOPE_AGENT);
            __hip_atomic_store(gen, g + 1u, __ATOMIC_RELEASE, __HIP_MEMORY_SCOPE_AGENT);
        } else {
            while (__hip_atomic_load(gen, __ATOMIC_ACQUIRE, __HIP_MEMORY_SCOPE_AGENT) == g)
                __builtin_amdgcn_s_sleep(2);
        }
    }
    __syncthreads();
}

// One-time per call: fp32->fp16 weight conversion, zero h/c state,
// initialize y with bl (atomic partial sums land on top), reset barrier.
__global__ void lstm_init_kernel(const float* __restrict__ Whh0,
                                 const float* __restrict__ Wih1,
                                 const float* __restrict__ Whh1,
                                 const float* __restrict__ bl,
                                 half_t* Whh0h, half_t* Wih1h, half_t* Whh1h,
                                 half_t* hbufs, float* cbufs, float* y,
                                 unsigned* sync_) {
    size_t i = (size_t)blockIdx.x * blockDim.x + threadIdx.x;
    size_t stride = (size_t)gridDim.x * blockDim.x;
    const size_t NW = (size_t)4 * Hdim * Hdim;
    for (size_t k = i; k < NW; k += stride) {
        Whh0h[k] = (half_t)Whh0[k];
        Wih1h[k] = (half_t)Wih1[k];
        Whh1h[k] = (half_t)Whh1[k];
    }
    const size_t NH = (size_t)4 * Bsz * Hdim;          // h0[2] + h1[2]
    for (size_t k = i; k < NH; k += stride) hbufs[k] = (half_t)0.0f;
    const size_t NC = (size_t)2 * Bsz * Hdim;          // c0 + c1
    for (size_t k = i; k < NC; k += stride) cbufs[k] = 0.0f;
    const size_t NY = (size_t)Bsz * Tlen;
    const float b = bl[0];
    for (size_t k = i; k < NY; k += stride) y[k] = b;
    if (i == 0) { sync_[0] = 0u; sync_[1] = 0u; }
}

__global__ __launch_bounds__(256, 1) void lstm_persistent(
    const float* __restrict__ x,
    const float* __restrict__ Wih0,
    const float* __restrict__ bih0, const float* __restrict__ bhh0,
    const float* __restrict__ bih1, const float* __restrict__ bhh1,
    const float* __restrict__ Wl,
    const half_t* __restrict__ Whh0h,
    const half_t* __restrict__ Wih1h,
    const half_t* __restrict__ Whh1h,
    half_t* h0buf, half_t* h1buf,
    float* c0, float* c1,
    unsigned* sync_, float* y) {
    extern __shared__ half_t Atile[];      // 32 rows x LDS_ROW halves (66KB)

    const int mpair = blockIdx.x >> 3;                        // 0..15
    const int wave  = threadIdx.x >> 5;                       // 0..7
    const int htile = ((blockIdx.x & 7) << 3) | wave;         // 0..63
    const int lane  = threadIdx.x & 31;
    const int hi    = lane >> 4;
    const int mbase = mpair << 5;                             // 32 rows
    const int lm    = hi << 3;                                // +r -> local M
    const int hc    = (htile << 4) | (lane & 15);             // hidden column
    unsigned* cnt = sync_;
    unsigned* gen = sync_ + 1;
    const v8f vzero = {};

    for (int t = 0; t < Tlen; ++t) {
        const half_t* h0cur = h0buf + (size_t)(t & 1) * (Bsz * Hdim);
        half_t*       h0nxt = h0buf + (size_t)((t + 1) & 1) * (Bsz * Hdim);
        const half_t* h1cur = h1buf + (size_t)(t & 1) * (Bsz * Hdim);
        half_t*       h1nxt = h1buf + (size_t)((t + 1) & 1) * (Bsz * Hdim);

        // ---------------- layer 0: gates = h0 @ Whh0^T (+ tiny x-term) -----
        __syncthreads();
        stage_rows32(Atile, h0cur, mbase);
        stage_wait();

        v8f acc[8];
#pragma unroll
        for (int g = 0; g < 8; ++g) acc[g] = vzero;
        gemm8(acc, Atile, Whh0h, htile);

        {
            F8 fg[8];
#pragma unroll
            for (int g = 0; g < 8; ++g) fg[g].v = acc[g];
#pragma unroll
            for (int mt = 0; mt < 2; ++mt) {
#pragma unroll
                for (int r = 0; r < 8; ++r) {
                    const int m = mbase + (mt << 4) + lm + r;
                    const float x0 = x[((size_t)m * Tlen + t) * 2 + 0];
                    const float x1 = x[((size_t)m * Tlen + t) * 2 + 1];
                    float gv[4];
#pragma unroll
                    for (int g = 0; g < 4; ++g) {
                        const int gi = (g << 10) + hc;
                        gv[g] = fg[mt * 4 + g].f[r] + bih0[gi] + bhh0[gi]
                              + x0 * Wih0[gi * 2 + 0] + x1 * Wih0[gi * 2 + 1];
                    }
                    const float ii = sigmoidf_(gv[0]);
                    const float ff = sigmoidf_(gv[1]);
                    const float gg = tanhf(gv[2]);
                    const float oo = sigmoidf_(gv[3]);
                    const size_t idx = (size_t)m * Hdim + hc;
                    const float c = ff * c0[idx] + ii * gg;
                    c0[idx] = c;
                    h0nxt[idx] = (half_t)(oo * tanhf(c));
                }
            }
        }
        grid_sync(cnt, gen);   // make full h0_new visible device-wide

        // ---------------- layer 1: gates = h0_new @ Wih1^T + h1 @ Whh1^T ---
        __syncthreads();
        stage_rows32(Atile, h0nxt, mbase);
        stage_wait();
#pragma unroll
        for (int g = 0; g < 8; ++g) acc[g] = vzero;
        gemm8(acc, Atile, Wih1h, htile);

        __syncthreads();
        stage_rows32(Atile, h1cur, mbase);
        stage_wait();
        gemm8(acc, Atile, Whh1h, htile);

        {
            F8 fg[8];
#pragma unroll
            for (int g = 0; g < 8; ++g) fg[g].v = acc[g];
            const float wl = Wl[hc];
#pragma unroll
            for (int mt = 0; mt < 2; ++mt) {
#pragma unroll
                for (int r = 0; r < 8; ++r) {
                    const int m = mbase + (mt << 4) + lm + r;
                    float gv[4];
#pragma unroll
                    for (int g = 0; g < 4; ++g) {
                        const int gi = (g << 10) + hc;
                        gv[g] = fg[mt * 4 + g].f[r] + bih1[gi] + bhh1[gi];
                    }
                    const float ii = sigmoidf_(gv[0]);
                    const float ff = sigmoidf_(gv[1]);
                    const float gg = tanhf(gv[2]);
                    const float oo = sigmoidf_(gv[3]);
                    const size_t idx = (size_t)m * Hdim + hc;
                    const float c = ff * c1[idx] + ii * gg;
                    c1[idx] = c;
                    const float hv = oo * tanhf(c);
                    h1nxt[idx] = (half_t)hv;

                    // y[m,t] partial: reduce the 16 lanes holding row m
                    float part = hv * wl;
                    part += __shfl_xor(part, 1, 16);
                    part += __shfl_xor(part, 2, 16);
                    part += __shfl_xor(part, 4, 16);
                    part += __shfl_xor(part, 8, 16);
                    if ((lane & 15) == 0)
                        atomicAdd(&y[(size_t)m * Tlen + t], part);
                }
            }
        }
        grid_sync(cnt, gen);   // h1_new + y visible before next step
    }
}

extern "C" void kernel_launch(void* const* d_in, const int* in_sizes, int n_in,
                              void* d_out, int out_size, void* d_ws, size_t ws_size,
                              hipStream_t stream) {
    const float* x    = (const float*)d_in[0];
    const float* Wih0 = (const float*)d_in[1];
    const float* Whh0 = (const float*)d_in[2];
    const float* bih0 = (const float*)d_in[3];
    const float* bhh0 = (const float*)d_in[4];
    const float* Wih1 = (const float*)d_in[5];
    const float* Whh1 = (const float*)d_in[6];
    const float* bih1 = (const float*)d_in[7];
    const float* bhh1 = (const float*)d_in[8];
    const float* Wl   = (const float*)d_in[9];
    const float* bl   = (const float*)d_in[10];
    (void)in_sizes; (void)n_in; (void)out_size; (void)ws_size;

    char* ws = (char*)d_ws;
    half_t* Whh0h = (half_t*)(ws + 0);              //  8 MB  fp16 Whh0
    half_t* Wih1h = (half_t*)(ws + 8388608);        //  8 MB  fp16 Wih1
    half_t* Whh1h = (half_t*)(ws + 16777216);       //  8 MB  fp16 Whh1
    half_t* hbufs = (half_t*)(ws + 25165824);       //  4 MB  h0[2], h1[2]
    float*  cbufs = (float* )(ws + 29360128);       //  4 MB  c0, c1
    unsigned* syn = (unsigned*)(ws + 33554432);     //  8 B   barrier state
    float* y = (float*)d_out;

    hipLaunchKernelGGL(lstm_init_kernel, dim3(2048), dim3(256), 0, stream,
                       Whh0, Wih1, Whh1, bl,
                       Whh0h, Wih1h, Whh1h,
                       hbufs, cbufs, y, syn);

    hipLaunchKernelGGL(lstm_persistent, dim3(NBLK), dim3(256), LDS_BYTES, stream,
                       x, Wih0, bih0, bhh0, bih1, bhh1, Wl,
                       Whh0h, Wih1h, Whh1h,
                       hbufs,                          // h0 double buffer
                       hbufs + (size_t)2 * Bsz * Hdim, // h1 double buffer
                       cbufs,                          // c0
                       cbufs + (size_t)Bsz * Hdim,     // c1
                       syn, y);
}
